// TransformerBlock_80049600463169
// MI455X (gfx1250) — compile-verified, matrix operand order checked by reading, not run on hardware
//
#include <hip/hip_runtime.h>
#include <math.h>

#define BATCH 4
#define SEQ   2048
#define DMODEL 1024
#define NHEAD 16
#define HD    64
#define MROWS (BATCH * SEQ)   // 8192

typedef __attribute__((ext_vector_type(16))) __bf16 v16bf;
typedef __attribute__((ext_vector_type(8)))  __bf16 v8bf;
typedef __attribute__((ext_vector_type(8)))  float  v8f;

// Build a 16x32 (or 32x16) bf16 WMMA fragment from two contiguous 8-element chunks.
__device__ __forceinline__ v16bf mkfrag(const __bf16* lo, const __bf16* hi) {
    v8bf a = *(const v8bf*)lo;
    v8bf b = *(const v8bf*)hi;
    v16bf r;
#pragma unroll
    for (int i = 0; i < 8; ++i) { r[i] = a[i]; r[i + 8] = b[i]; }
    return r;
}

__device__ __forceinline__ v8f wmma_bf16(v16bf a, v16bf b, v8f c) {
    return __builtin_amdgcn_wmma_f32_16x16x32_bf16(false, a, false, b, (short)0, c, false, false);
}

// CDNA5 async global->LDS copy (ASYNCcnt path, GLOBAL_LOAD_ASYNC_TO_LDS_B128).
__device__ __forceinline__ void async_g2l_b128(unsigned lds_off, const void* g) {
    asm volatile("global_load_async_to_lds_b128 %0, %1, off"
                 :: "v"(lds_off), "v"((unsigned long long)(uintptr_t)g)
                 : "memory");
}
__device__ __forceinline__ void wait_async0() {
    asm volatile("s_wait_asynccnt 0x0" ::: "memory");
}
__device__ __forceinline__ void wait_async5() {   // tile t done, tile t+1 may be in flight
    asm volatile("s_wait_asynccnt 0x5" ::: "memory");
}

// ---------------- weight fp32 [K][N] -> bf16 transposed [N][K], LDS-tiled ----------------
__global__ __launch_bounds__(256) void cvt_w_T_tiled(const float* __restrict__ w,
                                                     __bf16* __restrict__ wt, int K, int N) {
    __shared__ float tile[32][33];
    const int k0 = blockIdx.y * 32, n0 = blockIdx.x * 32;
    const int tx = threadIdx.x & 31, ty = threadIdx.x >> 5;   // 32 x 8
#pragma unroll
    for (int i = 0; i < 4; ++i) {
        int r = ty + i * 8;
        tile[r][tx] = w[(size_t)(k0 + r) * N + n0 + tx];      // coalesced read
    }
    __syncthreads();
#pragma unroll
    for (int i = 0; i < 4; ++i) {
        int r = ty + i * 8;
        wt[(size_t)(n0 + r) * K + k0 + tx] = (__bf16)tile[tx][r];   // coalesced write
    }
}

// ---------------- bf16 transpose per (b,h): v [BH][SEQ][HD] -> vt [BH][HD][SEQ] ----------------
__global__ __launch_bounds__(256) void transpose_v(const __bf16* __restrict__ v,
                                                   __bf16* __restrict__ vt) {
    __shared__ __bf16 tile[32][33];
    const int bh = blockIdx.z;
    const int l0 = blockIdx.x * 32, d0 = blockIdx.y * 32;
    const int tx = threadIdx.x & 31, ty = threadIdx.x >> 5;
    const __bf16* src = v + (size_t)bh * SEQ * HD;
    __bf16* dst = vt + (size_t)bh * HD * SEQ;
#pragma unroll
    for (int i = 0; i < 4; ++i) {
        int r = ty + i * 8;
        tile[r][tx] = src[(size_t)(l0 + r) * HD + d0 + tx];   // coalesced read
    }
    __syncthreads();
#pragma unroll
    for (int i = 0; i < 4; ++i) {
        int r = ty + i * 8;
        dst[(size_t)(d0 + r) * SEQ + l0 + tx] = tile[tx][r];  // coalesced write
    }
}

// ---------------- RMSNorm: fp32 row -> bf16 row ----------------
__global__ __launch_bounds__(256) void rmsnorm_bf16(const float* __restrict__ x,
                                                    const float* __restrict__ scale,
                                                    __bf16* __restrict__ h) {
    __shared__ float red[256];
    int row = blockIdx.x;
    const float* xr = x + (size_t)row * DMODEL;
    float s = 0.f;
#pragma unroll
    for (int i = 0; i < DMODEL / 256; ++i) {
        float v = xr[threadIdx.x + i * 256];
        s += v * v;
    }
    red[threadIdx.x] = s;
    __syncthreads();
    for (int off = 128; off > 0; off >>= 1) {
        if (threadIdx.x < off) red[threadIdx.x] += red[threadIdx.x + off];
        __syncthreads();
    }
    float rs = rsqrtf(red[0] / (float)DMODEL + 1e-8f);
    __bf16* hr = h + (size_t)row * DMODEL;
#pragma unroll
    for (int i = 0; i < DMODEL / 256; ++i) {
        int c = threadIdx.x + i * 256;
        hr[c] = (__bf16)(xr[c] * rs * scale[c]);
    }
}

// ---------------- bf16 GEMM: A[M,K] x Bt[N,K]^T, block 64x256, wave 32x64 ----------------
// Double-buffered async global->LDS pipeline: tile t+1 streams while tile t computes.
// EPI 0: bf16 out.  EPI 1: exact GELU -> bf16 out.  EPI 2: residual add -> fp32 out.
// EPI 3: QKV split: q -> outB, k -> outB2, v -> outB3, all [B,H,L,hd] (contiguous stores)
template <int EPI>
__global__ __launch_bounds__(256) void gemm_bf16(const __bf16* __restrict__ A,
                                                 const __bf16* __restrict__ Bt,
                                                 const float* __restrict__ resid,
                                                 float* __restrict__ outF,
                                                 __bf16* __restrict__ outB,
                                                 __bf16* __restrict__ outB2,
                                                 __bf16* __restrict__ outB3,
                                                 int M, int N, int K) {
    __shared__ __attribute__((aligned(16))) __bf16 As[2][64 * 32];
    __shared__ __attribute__((aligned(16))) __bf16 Bs[2][256 * 32];

    const int tid = threadIdx.x;
    const int w = tid >> 5, lane = tid & 31;
    const int hl = lane >> 4, r16 = lane & 15;
    const int waveM = w >> 2, waveN = w & 3;   // 2 x 4 wave grid
    const int m0 = blockIdx.y * 64, n0 = blockIdx.x * 256;

    v8f acc[2][4];
#pragma unroll
    for (int i = 0; i < 2; ++i)
#pragma unroll
        for (int j = 0; j < 4; ++j)
#pragma unroll
            for (int r = 0; r < 8; ++r) acc[i][j][r] = 0.f;

    const int rowt = tid >> 2, cc = (tid & 3) * 8;
    unsigned asB[2] = {(unsigned)(uintptr_t)&As[0][0], (unsigned)(uintptr_t)&As[1][0]};
    unsigned bsB[2] = {(unsigned)(uintptr_t)&Bs[0][0], (unsigned)(uintptr_t)&Bs[1][0]};

    auto stage = [&](int k0, int buf) {
        // A tile 64x32: 1 x b128 per thread; Bt tile 256x32: 4 x b128 per thread
        async_g2l_b128(asB[buf] + (unsigned)(rowt * 32 + cc) * 2,
                       A + (size_t)(m0 + rowt) * K + k0 + cc);
#pragma unroll
        for (int i = 0; i < 4; ++i) {
            int idx = tid + i * 256;
            int rowb = idx >> 2, cb = (idx & 3) * 8;
            async_g2l_b128(bsB[buf] + (unsigned)(rowb * 32 + cb) * 2,
                           Bt + (size_t)(n0 + rowb) * K + k0 + cb);
        }
    };

    const int T = K / 32;
    stage(0, 0);
    for (int t = 0; t < T; ++t) {
        const int buf = t & 1;
        if (t + 1 < T) { stage((t + 1) * 32, (t + 1) & 1); wait_async5(); }
        else           { wait_async0(); }
        __syncthreads();   // tile t resident for all waves

        v16bf af[2], bfr[4];
#pragma unroll
        for (int mt = 0; mt < 2; ++mt) {
            const __bf16* p = &As[buf][(waveM * 32 + mt * 16 + r16) * 32];
            af[mt] = mkfrag(p + hl * 8, p + 16 + hl * 8);
        }
#pragma unroll
        for (int nt = 0; nt < 4; ++nt) {
            const __bf16* p = &Bs[buf][(waveN * 64 + nt * 16 + r16) * 32];
            bfr[nt] = mkfrag(p + hl * 8, p + 16 + hl * 8);
        }
#pragma unroll
        for (int mt = 0; mt < 2; ++mt)
#pragma unroll
            for (int nt = 0; nt < 4; ++nt)
                acc[mt][nt] = wmma_bf16(af[mt], bfr[nt], acc[mt][nt]);
        __syncthreads();   // protect buf from being restaged while still being read
    }

#pragma unroll
    for (int mt = 0; mt < 2; ++mt)
#pragma unroll
        for (int nt = 0; nt < 4; ++nt)
#pragma unroll
            for (int r = 0; r < 8; ++r) {
                int row = m0 + waveM * 32 + mt * 16 + r + hl * 8;
                int col = n0 + waveN * 64 + nt * 16 + r16;
                size_t o = (size_t)row * N + col;
                float v = acc[mt][nt][r];
                if (EPI == 0) {
                    outB[o] = (__bf16)v;
                } else if (EPI == 1) {
                    outB[o] = (__bf16)(0.5f * v * (1.f + erff(v * 0.70710678118654752f)));
                } else if (EPI == 2) {
                    outF[o] = resid[o] + v;
                } else {
                    // split QKV into per-head layout [B,H,L,hd]; all stores lane-contiguous
                    int b = row >> 11, l = row & 2047;
                    int which = col >> 10, rem = col & 1023;
                    int h = rem >> 6, d = rem & 63;
                    size_t dstIdx = (((size_t)b * NHEAD + h) * SEQ + l) * HD + d;
                    __bf16 val = (__bf16)v;
                    __bf16* dst = (which == 0) ? outB : (which == 1) ? outB2 : outB3;
                    dst[dstIdx] = val;
                }
            }
}

// ---------------- flash attention: 1 wave = 16 query rows, 64-key blocks ----------------
__global__ __launch_bounds__(256) void attn_kernel(const __bf16* __restrict__ q,
                                                   const __bf16* __restrict__ k,
                                                   const __bf16* __restrict__ vt,
                                                   __bf16* __restrict__ out) {
    __shared__ __attribute__((aligned(16))) __bf16 pbuf[8][16][64];
    const int tid = threadIdx.x;
    const int w = tid >> 5, lane = tid & 31;
    const int hl = lane >> 4, r16 = lane & 15;
    const int bh = blockIdx.y;                 // b*NHEAD + h
    const int qbase = blockIdx.x * 128 + w * 16;

    const __bf16* qp = q + ((size_t)bh * SEQ + qbase) * HD;
    const __bf16* kp = k + (size_t)bh * SEQ * HD;
    const __bf16* vp = vt + (size_t)bh * HD * SEQ;

    // Q fragments for hd chunks [0,32) and [32,64), resident for whole kernel
    v16bf qa[2];
#pragma unroll
    for (int kk = 0; kk < 2; ++kk) {
        const __bf16* p = qp + (size_t)r16 * HD + kk * 32;
        qa[kk] = mkfrag(p + hl * 8, p + 16 + hl * 8);
    }
    // all-ones B fragment for WMMA row-sums
    v16bf ones;
#pragma unroll
    for (int i = 0; i < 16; ++i) ones[i] = (__bf16)1.0f;

    float m[8], lsum[8], al8[8];
    v8f o[4];
#pragma unroll
    for (int r = 0; r < 8; ++r) { m[r] = -INFINITY; lsum[r] = 0.f; }
#pragma unroll
    for (int nt = 0; nt < 4; ++nt)
#pragma unroll
        for (int r = 0; r < 8; ++r) o[nt][r] = 0.f;

    for (int jb = 0; jb < SEQ; jb += 64) {
        v8f s[4];
#pragma unroll
        for (int nt = 0; nt < 4; ++nt)
#pragma unroll
            for (int r = 0; r < 8; ++r) s[nt][r] = 0.f;

        // S = Q K^T for 64 keys (8 WMMA)
#pragma unroll
        for (int kk = 0; kk < 2; ++kk) {
#pragma unroll
            for (int nt = 0; nt < 4; ++nt) {
                const __bf16* p = kp + (size_t)(jb + nt * 16 + r16) * HD + kk * 32;
                v16bf kbf = mkfrag(p + hl * 8, p + 16 + hl * 8);
                s[nt] = wmma_bf16(qa[kk], kbf, s[nt]);
            }
        }
#pragma unroll
        for (int nt = 0; nt < 4; ++nt) s[nt] = s[nt] * 0.125f;   // hd^-0.5

        // online softmax: block max via one butterfly per row; P staged to LDS
#pragma unroll
        for (int r = 0; r < 8; ++r) {
            float mc = fmaxf(fmaxf(s[0][r], s[1][r]), fmaxf(s[2][r], s[3][r]));
#pragma unroll
            for (int msk = 1; msk < 16; msk <<= 1) mc = fmaxf(mc, __shfl_xor(mc, msk, 32));
            float mn = fmaxf(m[r], mc);
            float al = __expf(m[r] - mn);
            m[r] = mn;
            al8[r] = al;
#pragma unroll
            for (int nt = 0; nt < 4; ++nt) {
                float p = __expf(s[nt][r] - mn);
                pbuf[w][r + hl * 8][nt * 16 + r16] = (__bf16)p;  // C-layout -> A-layout
            }
#pragma unroll
            for (int nt = 0; nt < 4; ++nt) o[nt][r] *= al;
        }
        __syncthreads();

        v16bf pa0 = mkfrag(&pbuf[w][r16][hl * 8],      &pbuf[w][r16][16 + hl * 8]);
        v16bf pa1 = mkfrag(&pbuf[w][r16][32 + hl * 8], &pbuf[w][r16][48 + hl * 8]);

        // row sums of P via WMMA against all-ones B (replaces 32 ds_bpermutes)
        v8f rows;
#pragma unroll
        for (int r = 0; r < 8; ++r) rows[r] = 0.f;
        rows = wmma_bf16(pa0, ones, rows);
        rows = wmma_bf16(pa1, ones, rows);
#pragma unroll
        for (int r = 0; r < 8; ++r) lsum[r] = lsum[r] * al8[r] + rows[r];

        // O += P V  (8 WMMA)
#pragma unroll
        for (int nt = 0; nt < 4; ++nt) {
            const __bf16* p0 = vp + (size_t)(nt * 16 + r16) * SEQ + jb;
            v16bf vb0 = mkfrag(p0 + hl * 8, p0 + 16 + hl * 8);
            o[nt] = wmma_bf16(pa0, vb0, o[nt]);
            const __bf16* p1 = p0 + 32;
            v16bf vb1 = mkfrag(p1 + hl * 8, p1 + 16 + hl * 8);
            o[nt] = wmma_bf16(pa1, vb1, o[nt]);
        }
        __syncthreads();
    }

    const int b = bh >> 4, h = bh & 15;
#pragma unroll
    for (int nt = 0; nt < 4; ++nt)
#pragma unroll
        for (int r = 0; r < 8; ++r) {
            int row = qbase + r + hl * 8;
            out[((size_t)b * SEQ + row) * DMODEL + h * HD + nt * 16 + r16] =
                (__bf16)(o[nt][r] / lsum[r]);
        }
}

// ---------------- host-side orchestration ----------------
extern "C" void kernel_launch(void* const* d_in, const int* in_sizes, int n_in,
                              void* d_out, int out_size, void* d_ws, size_t ws_size,
                              hipStream_t stream) {
    const float* x     = (const float*)d_in[0];
    const float* n1s   = (const float*)d_in[1];
    const float* wqkv  = (const float*)d_in[2];
    const float* wproj = (const float*)d_in[3];
    const float* n2s   = (const float*)d_in[4];
    const float* wfc1  = (const float*)d_in[5];
    const float* wfc2  = (const float*)d_in[6];

    char* ws = (char*)d_ws;
    size_t cur = 0;
    auto alloc = [&](size_t bytes) -> char* {
        char* p = ws + cur;
        cur += (bytes + 255) & ~(size_t)255;
        return p;
    };

    __bf16* wqkvT = (__bf16*)alloc((size_t)3 * DMODEL * DMODEL * 2);
    __bf16* wprojT = (__bf16*)alloc((size_t)DMODEL * DMODEL * 2);
    __bf16* wfc1T = (__bf16*)alloc((size_t)4 * DMODEL * DMODEL * 2);
    __bf16* wfc2T = (__bf16*)alloc((size_t)4 * DMODEL * DMODEL * 2);
    __bf16* hbuf  = (__bf16*)alloc((size_t)MROWS * DMODEL * 2);
    __bf16* qb    = (__bf16*)alloc((size_t)MROWS * DMODEL * 2);
    __bf16* kbuf  = (__bf16*)alloc((size_t)MROWS * DMODEL * 2);
    __bf16* vbuf  = (__bf16*)alloc((size_t)MROWS * DMODEL * 2);
    __bf16* vtb   = (__bf16*)alloc((size_t)MROWS * DMODEL * 2);
    __bf16* attnb = (__bf16*)alloc((size_t)MROWS * DMODEL * 2);
    float*  ybuf  = (float*)alloc((size_t)MROWS * DMODEL * 4);
    __bf16* h2buf = (__bf16*)alloc((size_t)MROWS * DMODEL * 2);
    __bf16* gbuf  = (__bf16*)alloc((size_t)MROWS * 4 * DMODEL * 2);

    // 1) weight convert + transpose to bf16 [N][K] (LDS-tiled, coalesced both sides)
    cvt_w_T_tiled<<<dim3(3 * DMODEL / 32, DMODEL / 32), 256, 0, stream>>>(wqkv, wqkvT, DMODEL, 3 * DMODEL);
    cvt_w_T_tiled<<<dim3(DMODEL / 32, DMODEL / 32), 256, 0, stream>>>(wproj, wprojT, DMODEL, DMODEL);
    cvt_w_T_tiled<<<dim3(4 * DMODEL / 32, DMODEL / 32), 256, 0, stream>>>(wfc1, wfc1T, DMODEL, 4 * DMODEL);
    cvt_w_T_tiled<<<dim3(DMODEL / 32, 4 * DMODEL / 32), 256, 0, stream>>>(wfc2, wfc2T, 4 * DMODEL, DMODEL);

    // 2) rmsnorm1 -> bf16
    rmsnorm_bf16<<<MROWS, 256, 0, stream>>>(x, n1s, hbuf);

    // 3) QKV GEMM with fused per-head q/k/v split epilogue (contiguous stores)
    gemm_bf16<3><<<dim3(3 * DMODEL / 256, MROWS / 64), 256, 0, stream>>>(
        hbuf, wqkvT, nullptr, nullptr, qb, kbuf, vbuf, MROWS, 3 * DMODEL, DMODEL);

    // 4) v -> v^T per head (LDS-tiled, coalesced)
    transpose_v<<<dim3(SEQ / 32, HD / 32, BATCH * NHEAD), 256, 0, stream>>>(vbuf, vtb);

    // 5) flash attention
    attn_kernel<<<dim3(SEQ / 128, BATCH * NHEAD), 256, 0, stream>>>(qb, kbuf, vtb, attnb);

    // 6) proj GEMM + residual (fp32 out)
    gemm_bf16<2><<<dim3(DMODEL / 256, MROWS / 64), 256, 0, stream>>>(
        attnb, wprojT, x, ybuf, nullptr, nullptr, nullptr, MROWS, DMODEL, DMODEL);

    // 7) rmsnorm2 -> bf16
    rmsnorm_bf16<<<MROWS, 256, 0, stream>>>(ybuf, n2s, h2buf);

    // 8) FC1 GEMM + exact GELU (bf16 out)
    gemm_bf16<1><<<dim3(4 * DMODEL / 256, MROWS / 64), 256, 0, stream>>>(
        h2buf, wfc1T, nullptr, nullptr, gbuf, nullptr, nullptr, MROWS, 4 * DMODEL, DMODEL);

    // 9) FC2 GEMM + residual -> final fp32 output
    gemm_bf16<2><<<dim3(DMODEL / 256, MROWS / 64), 256, 0, stream>>>(
        gbuf, wfc2T, ybuf, (float*)d_out, nullptr, nullptr, nullptr, MROWS, DMODEL, 4 * DMODEL);

    (void)in_sizes; (void)n_in; (void)out_size; (void)ws_size;
}
